// UpXiPredictor_9182640078991
// MI455X (gfx1250) — matrix operand order repaired and checked by reading, not compiled, not run on hardware
//
#include <hip/hip_runtime.h>

// ---------------------------------------------------------------------------
// UpXiPredictor fused kernel for gfx1250 (MI455X, wave32, WMMA + async-LDS)
//
// Per workgroup (32 node-rows, 256 threads = 8 wave32):
//   stage P : per-row gather index / mask / pos01, Fourier features
//   stage G : async DMA gather of s_parent rows -> LDS fp32
//             (global_load_async_to_lds_b128, ASYNCcnt)
//   stage 0 : s_pi = gathered + (feats @ pos_w^T + pos_b)*mask -> LDS bf16
//   LN(q)   : in-place LayerNorm (4 threads/row, shfl_xor reduce)
//   L1      : 384x384 bf16 WMMA GEMM + SiLU            (actA -> actB)
//   L2      : 384x384 bf16 WMMA GEMM + mask            (actB -> actA)
//   LN(m)   : in-place LayerNorm
//   L3      : 384->256 bf16 WMMA GEMM + SiLU           (actA -> actB)
//   L4      : 256->256 bf16 WMMA GEMM + SiLU           (actB -> actA)
//   head    : 256->3 scalar dots, tanh*1.5, rigid apply, outputs
// ---------------------------------------------------------------------------

#define ROWS    32            // node rows per workgroup
#define LDA     392           // bf16 stride (mult of 8 -> 16B aligned rows)
#define NNODES  65536         // B*N
#define NSHIFT  14            // N = 16384
#define KPAR    2048
#define CDIM    384
#define HDIM    256

// LDS layout (byte offsets within the dynamic allocation; gather buffer first
// so the async-LDS inline asm uses exact offsets from the allocation base)
#define GATHER_BYTES  (ROWS * CDIM * 4)               // 49152: fp32 gather stage
#define ACTA_OFF      (GATHER_BYTES)                  // ROWS*LDA bf16
#define ACTB_OFF      (ACTA_OFF + ROWS * LDA * 2)
#define FEATS_OFF     (ACTB_OFF + ROWS * LDA * 2)     // ROWS*32 f32
#define MASK_OFF      (FEATS_OFF + ROWS * 32 * 4)
#define IDX_OFF       (MASK_OFF + ROWS * 4)
#define SMEM_BYTES    (IDX_OFF + ROWS * 4)            // 103680 B

typedef __attribute__((ext_vector_type(16))) __bf16 v16bf;
typedef __attribute__((ext_vector_type(8)))  float  v8f;

struct __align__(16) U4 { unsigned x, y, z, w; };
struct __align__(32) B32 { U4 lo, hi; };

__device__ __forceinline__ unsigned short f2bf(float f) {
    unsigned u = __builtin_bit_cast(unsigned, f);
    unsigned r = u + 0x7fffu + ((u >> 16) & 1u);   // round-to-nearest-even
    return (unsigned short)(r >> 16);
}
__device__ __forceinline__ float bf2f(unsigned short h) {
    unsigned u = ((unsigned)h) << 16;
    return __builtin_bit_cast(float, u);
}

// ---------------------------------------------------------------------------
// bf16 WMMA GEMM layer: out[ROWS][NOUT] = act(in[ROWS][KDIM] @ W^T + bias)
// W is row-major [NOUT][KDIM] bf16 in global (L2-resident).
// Wave w: row tile = w&1 (2 tiles of 16 rows), column tiles strided by 4.
// A fragment (16-bit A 16x32, ISA 7.12.2): lane holds M=lane&15,
//   K in {g*8..g*8+7} and {16+g*8..16+g*8+7}, g=lane>>4  -> two 16B loads.
// B fragment: lane holds N=lane&15, same K set as A       -> two 16B loads.
// D: M = v + 8*g, N = lane&15.
// ---------------------------------------------------------------------------
template<int KDIM, int NOUT, bool SILU, bool MASK>
__device__ __forceinline__ void gemm_layer(
    const unsigned short* __restrict__ W,
    const float* __restrict__ bias,
    const unsigned short* __restrict__ in,    // LDS bf16 [ROWS][LDA]
    unsigned short* __restrict__ out,         // LDS bf16 [ROWS][LDA]
    const float* __restrict__ maskL,
    int wave, int lane)
{
    const int rowtile = wave & 1;
    const int ngroup  = wave >> 1;            // 4 column-tile groups
    const int m = lane & 15;
    const int g = lane >> 4;
    const int NT = NOUT / 16;

    const unsigned short* arow = in + (rowtile * 16 + m) * LDA;

    for (int nt = ngroup; nt < NT; nt += 4) {
        const int ncol = nt * 16 + m;
        const unsigned short* wrow = W + (size_t)ncol * KDIM;
        if (nt + 4 < NT)  // warm L2/L0 for the next column tile (global_prefetch_b8)
            __builtin_prefetch(W + (size_t)(ncol + 64) * KDIM, 0, 0);

        v8f acc = {};
        #pragma unroll
        for (int kc = 0; kc < KDIM; kc += 32) {
            B32 ta, tb;
            const unsigned short* ap = arow + kc + g * 8;
            ta.lo = *(const U4*)(ap);
            ta.hi = *(const U4*)(ap + 16);
            const unsigned short* bp = wrow + kc + g * 8;
            tb.lo = *(const U4*)(bp);
            tb.hi = *(const U4*)(bp + 16);
            v16bf a = __builtin_bit_cast(v16bf, ta);
            v16bf b = __builtin_bit_cast(v16bf, tb);
            acc = __builtin_amdgcn_wmma_f32_16x16x32_bf16(
                false, a, false, b, (short)0, acc, false, false);
        }

        const float bv = bias[ncol];
        #pragma unroll
        for (int v = 0; v < 8; v++) {
            int row = rowtile * 16 + v + 8 * g;
            float val = acc[v] + bv;
            if (SILU) val = val / (1.0f + __expf(-val));   // x*sigmoid(x)
            if (MASK) val *= maskL[row];
            out[row * LDA + ncol] = f2bf(val);
        }
    }
}

// in-place LayerNorm over 384 cols, 4 threads per row
__device__ __forceinline__ void ln_rows(unsigned short* buf,
                                        const float* __restrict__ gw,
                                        const float* __restrict__ gb,
                                        int tid)
{
    if (tid < ROWS * 4) {
        const int r = tid >> 2, part = tid & 3;
        unsigned short* rp = buf + r * LDA;
        float s = 0.f, ss = 0.f;
        const int c0 = part * 96;
        for (int c = c0; c < c0 + 96; c++) {
            float v = bf2f(rp[c]); s += v; ss += v * v;
        }
        s  += __shfl_xor(s, 1, 32);  s  += __shfl_xor(s, 2, 32);
        ss += __shfl_xor(ss, 1, 32); ss += __shfl_xor(ss, 2, 32);
        const float mean = s * (1.0f / 384.0f);
        const float inv  = rsqrtf(ss * (1.0f / 384.0f) - mean * mean + 1e-5f);
        for (int c = c0; c < c0 + 96; c++) {
            float v = (bf2f(rp[c]) - mean) * inv * gw[c] + gb[c];
            rp[c] = f2bf(v);
        }
    }
}

__global__ void cvt_bf16_kernel(const float* __restrict__ src,
                                unsigned short* __restrict__ dst, int n)
{
    int i = blockIdx.x * blockDim.x + threadIdx.x;
    if (i < n) dst[i] = f2bf(src[i]);
}

__global__ __launch_bounds__(256)
void upxi_fused_kernel(
    const float* __restrict__ s_parent, const float* __restrict__ mu_k,
    const float* __restrict__ R_k,      const float* __restrict__ s_k,
    const int*   __restrict__ a_idx,    const float* __restrict__ node_mask,
    const float* __restrict__ pos01,
    const float* __restrict__ pos_w,    const float* __restrict__ pos_b,
    const float* __restrict__ q_ln_g,   const float* __restrict__ q_ln_b,
    const float* __restrict__ q_b1,     const float* __restrict__ q_b2,
    const float* __restrict__ m_ln_g,   const float* __restrict__ m_ln_b,
    const float* __restrict__ m_b1,     const float* __restrict__ m_b2,
    const float* __restrict__ m_w3,     const float* __restrict__ m_b3,
    const unsigned short* __restrict__ w1bf, const unsigned short* __restrict__ w2bf,
    const unsigned short* __restrict__ w3bf, const unsigned short* __restrict__ w4bf,
    float* __restrict__ out_xi, float* __restrict__ out_x, float* __restrict__ out_p)
{
    extern __shared__ unsigned char smem[];
    float*          gatherF = (float*)smem;                     // LDS offset 0
    unsigned short* actA    = (unsigned short*)(smem + ACTA_OFF);
    unsigned short* actB    = (unsigned short*)(smem + ACTB_OFF);
    float*          featsL  = (float*)(smem + FEATS_OFF);
    float*          maskL   = (float*)(smem + MASK_OFF);
    int*            idxL    = (int*)(smem + IDX_OFF);
    float*          xiL     = featsL;                           // alias (post-stage0)

    const int tid  = threadIdx.x;
    const int wave = tid >> 5;
    const int lane = tid & 31;
    const int row0 = blockIdx.x * ROWS;

    // ---- stage P: per-row setup + Fourier features + pos01 passthrough ----
    if (tid < ROWS) {
        const int r = tid;
        const int mg = row0 + r;
        int ii = a_idx[mg];
        ii = ii < 0 ? 0 : (ii > KPAR - 1 ? KPAR - 1 : ii);
        idxL[r]  = ii;
        maskL[r] = node_mask[mg];
        const float p = pos01[mg];
        out_p[mg] = p;
        const float f0 = 3.14159265358979323846f * p;
        #pragma unroll
        for (int j = 0; j < 16; j++) {
            const float ang = f0 * (float)(1 << j);
            featsL[r * 32 + j]      = __sinf(ang);
            featsL[r * 32 + 16 + j] = __cosf(ang);
        }
    }
    __syncthreads();

    // ---- stage G: async DMA gather of s_parent rows into LDS (fp32) ----
    // 96 x 16B chunks per row, ROWS*96 = 3072 chunks, 12 per thread.
    // global_load_async_to_lds_b128: VDST = LDS byte offset (LDS_BASE added
    // by HW), VADDR = 64-bit global address, tracked by ASYNCcnt.
    for (int t = tid; t < ROWS * 96; t += 256) {
        const int r  = t / 96;
        const int ck = t - r * 96;
        const int b  = (row0 + r) >> NSHIFT;
        const float* src = s_parent + (((size_t)b) * KPAR + idxL[r]) * CDIM + ck * 4;
        const unsigned      ldsoff = (unsigned)(r * (CDIM * 4) + ck * 16);
        const unsigned long long ga = (unsigned long long)(uintptr_t)src;
        asm volatile("global_load_async_to_lds_b128 %0, %1, off"
                     :: "v"(ldsoff), "v"(ga) : "memory");
    }
    asm volatile("s_wait_asynccnt 0x0" ::: "memory");
    __syncthreads();

    // ---- stage 0: gathered + positional linear -> actA (bf16) ----
    for (int t = tid; t < ROWS * CDIM; t += 256) {
        const int r = t / CDIM;
        const int c = t - r * CDIM;
        float acc = pos_b[c];
        const float* pw = pos_w + c * 32;
        const float* ft = featsL + r * 32;
        #pragma unroll
        for (int j = 0; j < 32; j++) acc = fmaf(ft[j], pw[j], acc);
        acc *= maskL[r];
        const float sv = gatherF[r * CDIM + c];
        actA[r * LDA + c] = f2bf(sv + acc);
    }
    __syncthreads();

    ln_rows(actA, q_ln_g, q_ln_b, tid);
    __syncthreads();
    gemm_layer<CDIM, CDIM, true,  false>(w1bf, q_b1, actA, actB, maskL, wave, lane);
    __syncthreads();
    gemm_layer<CDIM, CDIM, false, true >(w2bf, q_b2, actB, actA, maskL, wave, lane);
    __syncthreads();
    ln_rows(actA, m_ln_g, m_ln_b, tid);
    __syncthreads();
    gemm_layer<CDIM, HDIM, true,  false>(w3bf, m_b1, actA, actB, maskL, wave, lane);
    __syncthreads();
    gemm_layer<HDIM, HDIM, true,  false>(w4bf, m_b2, actB, actA, maskL, wave, lane);
    __syncthreads();

    // ---- head: 256->3, tanh*1.5 ----
    if (tid < ROWS * 3) {
        const int r = tid / 3, j = tid - r * 3;
        const unsigned short* h = actA + r * LDA;
        float acc = m_b3[j];
        const float* w = m_w3 + j * HDIM;
        #pragma unroll 8
        for (int k = 0; k < HDIM; k++) acc = fmaf(bf2f(h[k]), w[k], acc);
        acc *= maskL[r];
        const float xi = tanhf(acc) * 1.5f;
        xiL[r * 3 + j] = xi;
        out_xi[(size_t)(row0 + r) * 3 + j] = xi;
    }
    __syncthreads();

    // ---- rigid apply: x = R @ (xi * s) + mu ----
    if (tid < ROWS) {
        const int r = tid;
        const int mg = row0 + r;
        const int b = mg >> NSHIFT;
        const size_t base = ((size_t)b) * KPAR + idxL[r];
        const float t0 = xiL[r * 3 + 0] * fmaxf(s_k[base * 3 + 0], 1e-8f);
        const float t1 = xiL[r * 3 + 1] * fmaxf(s_k[base * 3 + 1], 1e-8f);
        const float t2 = xiL[r * 3 + 2] * fmaxf(s_k[base * 3 + 2], 1e-8f);
        const float* R = R_k + base * 9;
        const float mk10 = maskL[r] * 10.0f;
        #pragma unroll
        for (int i = 0; i < 3; i++) {
            float x = R[i * 3 + 0] * t0 + R[i * 3 + 1] * t1 + R[i * 3 + 2] * t2
                    + mu_k[base * 3 + i];
            out_x[(size_t)mg * 3 + i] = x * mk10;
        }
    }
}

extern "C" void kernel_launch(void* const* d_in, const int* in_sizes, int n_in,
                              void* d_out, int out_size, void* d_ws, size_t ws_size,
                              hipStream_t stream) {
    const float* s_parent  = (const float*)d_in[0];
    const float* mu_k      = (const float*)d_in[1];
    const float* R_k       = (const float*)d_in[2];
    const float* s_k       = (const float*)d_in[3];
    const int*   a_idx     = (const int*)  d_in[4];
    const float* node_mask = (const float*)d_in[5];
    const float* pos01     = (const float*)d_in[6];
    const float* pos_w     = (const float*)d_in[7];
    const float* pos_b     = (const float*)d_in[8];
    const float* q_ln_g    = (const float*)d_in[9];
    const float* q_ln_b    = (const float*)d_in[10];
    const float* q_w1      = (const float*)d_in[11];
    const float* q_b1      = (const float*)d_in[12];
    const float* q_w2      = (const float*)d_in[13];
    const float* q_b2      = (const float*)d_in[14];
    const float* m_ln_g    = (const float*)d_in[15];
    const float* m_ln_b    = (const float*)d_in[16];
    const float* m_w1      = (const float*)d_in[17];
    const float* m_b1      = (const float*)d_in[18];
    const float* m_w2      = (const float*)d_in[19];
    const float* m_b2      = (const float*)d_in[20];
    const float* m_w3      = (const float*)d_in[21];
    const float* m_b3      = (const float*)d_in[22];

    // bf16 weight staging in workspace (all 16B aligned)
    unsigned short* w1 = (unsigned short*)d_ws;        // 384*384
    unsigned short* w2 = w1 + CDIM * CDIM;             // 384*384
    unsigned short* w3 = w2 + CDIM * CDIM;             // 256*384
    unsigned short* w4 = w3 + HDIM * CDIM;             // 256*256

    cvt_bf16_kernel<<<dim3((CDIM * CDIM + 255) / 256), dim3(256), 0, stream>>>(q_w1, w1, CDIM * CDIM);
    cvt_bf16_kernel<<<dim3((CDIM * CDIM + 255) / 256), dim3(256), 0, stream>>>(q_w2, w2, CDIM * CDIM);
    cvt_bf16_kernel<<<dim3((HDIM * CDIM + 255) / 256), dim3(256), 0, stream>>>(m_w1, w3, HDIM * CDIM);
    cvt_bf16_kernel<<<dim3((HDIM * HDIM + 255) / 256), dim3(256), 0, stream>>>(m_w2, w4, HDIM * HDIM);

    float* out_xi = (float*)d_out;
    float* out_x  = out_xi + (size_t)NNODES * 3;
    float* out_p  = out_xi + (size_t)NNODES * 6;

    upxi_fused_kernel<<<dim3(NNODES / ROWS), dim3(256), SMEM_BYTES, stream>>>(
        s_parent, mu_k, R_k, s_k, a_idx, node_mask, pos01,
        pos_w, pos_b, q_ln_g, q_ln_b, q_b1, q_b2,
        m_ln_g, m_ln_b, m_b1, m_b2, m_w3, m_b3,
        w1, w2, w3, w4, out_xi, out_x, out_p);
}
